// GroupQueryAttention_3624952398050
// MI455X (gfx1250) — compile-verified, matrix operand order checked by reading, not run on hardware
//
#include <hip/hip_runtime.h>

// ---------- problem constants ----------
#define B_ 2
#define S_ 1024
#define HID_ 2048
#define HQ_ 32
#define HKV_ 8
#define HD_ 64
#define KVD_ (HKV_ * HD_)   // 512
#define M_ (B_ * S_)        // 2048
#define SCALE_ 0.125f       // 1/sqrt(64)
#define NEG_INF_ -1e30f

// ---------- WMMA types ----------
typedef __attribute__((ext_vector_type(16))) __bf16 bf16x16;
typedef __attribute__((ext_vector_type(8)))  __bf16 bf16x8;
typedef __attribute__((ext_vector_type(8)))  float  f32x8;
typedef __attribute__((ext_vector_type(4)))  unsigned int u32x4;
typedef __attribute__((ext_vector_type(8)))  int i32x8;
typedef __attribute__((ext_vector_type(4)))  int i32x4;

union BFrag { bf16x16 v; bf16x8 h[2]; };

static __device__ inline f32x8 wmma_bf16(bf16x16 a, bf16x16 b, f32x8 c) {
  return __builtin_amdgcn_wmma_f32_16x16x32_bf16(false, a, false, b, (short)0, c,
                                                 false, false);
}

static __device__ inline f32x8 zero8() {
  f32x8 z;
#pragma unroll
  for (int e = 0; e < 8; ++e) z[e] = 0.f;
  return z;
}

// ---------- Tensor Data Mover ----------
#if defined(__gfx1250__) && __has_builtin(__builtin_amdgcn_tensor_load_to_lds)
#define USE_TDM 1
#else
#define USE_TDM 0
#endif

#if USE_TDM
// Issue a 2D TDM tile load: tile_d0 x tile_d1 elements (2B each), contiguous
// along dim0 in both global (row stride = stride_elems) and LDS (packed rows).
static __device__ inline void tdm_load_2d_bf16(unsigned lds_off, const void* gaddr,
                                               unsigned tensor_d0, unsigned tensor_d1,
                                               unsigned tile_d0, unsigned tile_d1,
                                               unsigned long long stride_elems) {
  unsigned long long ga = (unsigned long long)(uintptr_t)gaddr;
  u32x4 g0;
  g0[0] = 1u;                                          // count=1 (valid), user mode
  g0[1] = lds_off;                                     // lds_addr
  g0[2] = (unsigned)(ga & 0xFFFFFFFFu);                // global_addr[31:0]
  g0[3] = (unsigned)((ga >> 32) & 0x01FFFFFFu) | (2u << 30);  // addr[56:32] | type=2
  i32x8 g1;
  g1[0] = (int)(1u << 16);                             // wg_mask=0, data_size=2B
  g1[1] = (int)(tensor_d0 << 16);                      // [63:48] = tensor_dim0 lo16
  g1[2] = (int)((tensor_d0 >> 16) | (tensor_d1 << 16));// dim0 hi16 | dim1 lo16
  g1[3] = (int)((tensor_d1 >> 16) | (tile_d0 << 16));  // dim1 hi16 | tile_dim0
  g1[4] = (int)(tile_d1 & 0xFFFFu);                    // tile_dim1, tile_dim2=0
  g1[5] = (int)(unsigned)(stride_elems & 0xFFFFFFFFull);     // dim0_stride lo32
  g1[6] = (int)(unsigned)((stride_elems >> 32) & 0xFFFFull); // dim0_stride hi16
  g1[7] = 0;                                           // dim1_stride (unused, 2D)
  i32x4 z4 = {0, 0, 0, 0};
#if __clang_major__ >= 23
  i32x8 z8 = {0, 0, 0, 0, 0, 0, 0, 0};
  __builtin_amdgcn_tensor_load_to_lds(g0, g1, z4, z4, z8, 0);
#else
  __builtin_amdgcn_tensor_load_to_lds(g0, g1, z4, z4, 0);
#endif
}

static __device__ inline unsigned lds_off_of(const void* p) {
  return (unsigned)(uintptr_t)p;   // generic LDS addr: low 32 bits = LDS offset
}
#endif

// ---------- elementwise f32 -> bf16 ----------
__global__ __launch_bounds__(256) void cvt_f32_bf16(const float* __restrict__ in,
                                                    __bf16* __restrict__ out, int n) {
  int i = blockIdx.x * 256 + threadIdx.x;
  if (i < n) out[i] = (__bf16)in[i];
}

// ---------- f32 (KxN) -> bf16 transposed (NxK) ----------
__global__ __launch_bounds__(256) void cvt_transpose_bf16(const float* __restrict__ in,
                                                          __bf16* __restrict__ out,
                                                          int K, int N) {
  int i = blockIdx.x * 256 + threadIdx.x;
  if (i >= K * N) return;
  int k = i / N, n = i - k * N;
  out[(size_t)n * K + k] = (__bf16)in[i];
}

// ---------- V re-layout: f32 (B,S,HKV,HD) -> bf16 (B,HKV,HD,S) ----------
__global__ __launch_bounds__(256) void cvt_v_transpose(const float* __restrict__ in,
                                                       __bf16* __restrict__ out,
                                                       int total) {
  int i = blockIdx.x * 256 + threadIdx.x;
  if (i >= total) return;
  int d = i & (HD_ - 1);
  int tmp = i >> 6;                   // (b*S + s)*HKV + h
  int h = tmp & (HKV_ - 1);
  int sh = tmp >> 3;                  // b*S + s
  int s = sh & (S_ - 1);
  int b = sh >> 10;
  out[(((size_t)(b * HKV_ + h) * HD_) + d) * S_ + s] = (__bf16)in[i];
}

// ---------- RoPE: f32 (B,S,H,HD) -> bf16 head-major (B,H,S,HD) ----------
__global__ __launch_bounds__(256) void rope_kernel(const float* __restrict__ in,
                                                   __bf16* __restrict__ out,
                                                   const float* __restrict__ cosT,
                                                   const float* __restrict__ sinT,
                                                   int total, int nheads) {
  int i = blockIdx.x * 256 + threadIdx.x;
  if (i >= total) return;
  int d = i & (HD_ - 1);
  int tmp = i >> 6;                   // (b*S + s)*H + h
  int h = tmp % nheads;
  int sh = tmp / nheads;              // b*S + s
  int s = sh & (S_ - 1);
  int b = sh >> 10;
  float x = in[i];
  float other = (d < HD_ / 2) ? -in[i + HD_ / 2] : in[i - HD_ / 2];
  float v = x * cosT[s * HD_ + d] + other * sinT[s * HD_ + d];
  out[(((size_t)(b * nheads + h) * S_) + s) * HD_ + d] = (__bf16)v;
}

// ---------- GEMM: C(f32, MxN) = A(bf16, MxK row-major) x Bt(bf16, NxK row-major)
#define BM 128
#define BN 128
#define BK 32
__global__ __launch_bounds__(256) void gemm_bf16_f32(const __bf16* __restrict__ A,
                                                     const __bf16* __restrict__ Bt,
                                                     float* __restrict__ C,
                                                     int M, int N, int K) {
  __shared__ __align__(16) __bf16 sA[2][BM][BK];   // row-major (m, k)
  __shared__ __align__(16) __bf16 sB[2][BN][BK];   // N-major  (n, k)
  const int tid  = threadIdx.x;
  const int lane = tid & 31;
  const int wid  = tid >> 5;
  const int wm   = wid >> 2;            // 0..1 : 64-row slab
  const int wn   = wid & 3;             // 0..3 : 32-col slab
  const int m0   = blockIdx.y * BM;
  const int n0   = blockIdx.x * BN;
  const int r    = lane & 15;
  const int hi   = lane >> 4;

  f32x8 acc[4][2];
#pragma unroll
  for (int i = 0; i < 4; ++i)
#pragma unroll
    for (int j = 0; j < 2; ++j) acc[i][j] = zero8();

  const int nk = K / BK;

#if USE_TDM
  if (tid < 32) {   // wave 0 is the TDM issuer
    tdm_load_2d_bf16(lds_off_of(&sA[0][0][0]), A + (size_t)m0 * K,
                     (unsigned)K, (unsigned)M, BK, BM, (unsigned long long)K);
    tdm_load_2d_bf16(lds_off_of(&sB[0][0][0]), Bt + (size_t)n0 * K,
                     (unsigned)K, (unsigned)N, BK, BN, (unsigned long long)K);
  }
#endif

  for (int it = 0; it < nk; ++it) {
    const int buf = it & 1;
    const int kt  = it * BK;
#if USE_TDM
    if (tid < 32) {
      if (it + 1 < nk) {
        const int ktn = kt + BK;
        tdm_load_2d_bf16(lds_off_of(&sA[buf ^ 1][0][0]), A + (size_t)m0 * K + ktn,
                         (unsigned)K, (unsigned)M, BK, BM, (unsigned long long)K);
        tdm_load_2d_bf16(lds_off_of(&sB[buf ^ 1][0][0]), Bt + (size_t)n0 * K + ktn,
                         (unsigned)K, (unsigned)N, BK, BN, (unsigned long long)K);
        __builtin_amdgcn_s_wait_tensorcnt(2);   // current pair complete (in-order)
      } else {
        __builtin_amdgcn_s_wait_tensorcnt(0);
      }
    }
#else
#pragma unroll
    for (int itc = 0; itc < 2; ++itc) {
      int c = tid + itc * 256;
      int row = c >> 2;
      int kk  = (c & 3) * 8;
      *(bf16x8*)&sA[buf][row][kk] =
          *(const bf16x8*)&A[(size_t)(m0 + row) * K + kt + kk];
      *(bf16x8*)&sB[buf][row][kk] =
          *(const bf16x8*)&Bt[(size_t)(n0 + row) * K + kt + kk];
    }
#endif
    __syncthreads();

    BFrag bfr[2];
#pragma unroll
    for (int j = 0; j < 2; ++j) {
      int col = wn * 32 + j * 16 + r;
      bfr[j].h[0] = *(const bf16x8*)&sB[buf][col][hi * 8];
      bfr[j].h[1] = *(const bf16x8*)&sB[buf][col][16 + hi * 8];
    }
#pragma unroll
    for (int i = 0; i < 4; ++i) {
      BFrag afr;
      int row = wm * 64 + i * 16 + r;
      afr.h[0] = *(const bf16x8*)&sA[buf][row][hi * 8];
      afr.h[1] = *(const bf16x8*)&sA[buf][row][16 + hi * 8];
#pragma unroll
      for (int j = 0; j < 2; ++j) acc[i][j] = wmma_bf16(afr.v, bfr[j].v, acc[i][j]);
    }
    __syncthreads();
  }

#pragma unroll
  for (int i = 0; i < 4; ++i)
#pragma unroll
    for (int j = 0; j < 2; ++j) {
      int col = n0 + wn * 32 + j * 16 + r;
#pragma unroll
      for (int e = 0; e < 8; ++e) {
        int row = m0 + wm * 64 + i * 16 + e + 8 * hi;
        C[(size_t)row * N + col] = acc[i][j][e];
      }
    }
}

// ---------- flash attention ----------
// Q: (B,HQ,S,HD) bf16; K: (B,HKV,S,HD) bf16; V: (B,HKV,HD,S) bf16 (transposed).
// grid: (S/64, HQ, B), block: 128 (4 waves); each wave owns 16 q rows.
// K/V chunks are plain 2D strided tiles -> staged by TDM, double-buffered.
__global__ __launch_bounds__(128) void attn_kernel(const __bf16* __restrict__ Q,
                                                   const __bf16* __restrict__ Kc,
                                                   const __bf16* __restrict__ Vc,
                                                   __bf16* __restrict__ O) {
  __shared__ __align__(16) __bf16 sK[2][32][HD_];  // kv-major (N-major for QK^T)
  __shared__ __align__(16) __bf16 sV[2][HD_][32];  // hd-major (N-major for PV)
  __shared__ __align__(16) __bf16 sP[4][16][32];   // per-wave P tile

  const int qblk = blockIdx.x;
  const int hq   = blockIdx.y;
  const int b    = blockIdx.z;
  const int hkv  = hq >> 2;                        // n_rep = 4
  const int tid  = threadIdx.x;
  const int lane = tid & 31;
  const int w    = tid >> 5;
  const int r    = lane & 15;
  const int hi   = lane >> 4;
  const int qbase = qblk * 64;
  const int wq    = qbase + w * 16;

  const __bf16* Kbase = Kc + ((size_t)(b * HKV_ + hkv) * S_) * HD_;
  const __bf16* Vbase = Vc + ((size_t)(b * HKV_ + hkv) * HD_) * S_;

  // Q fragments (hd split into two K=32 chunks)
  BFrag qa[2];
  {
    const __bf16* qrow = Q + (((size_t)(b * HQ_ + hq) * S_) + wq + r) * HD_;
    qa[0].h[0] = *(const bf16x8*)(qrow + hi * 8);
    qa[0].h[1] = *(const bf16x8*)(qrow + 16 + hi * 8);
    qa[1].h[0] = *(const bf16x8*)(qrow + 32 + hi * 8);
    qa[1].h[1] = *(const bf16x8*)(qrow + 48 + hi * 8);
  }

  f32x8 o[4];
#pragma unroll
  for (int ot = 0; ot < 4; ++ot) o[ot] = zero8();
  float m_[8], l_[8], alpha_[8];
#pragma unroll
  for (int e = 0; e < 8; ++e) { m_[e] = NEG_INF_; l_[e] = 0.f; }

  const int nchunks = (qbase + 64) / 32;   // causal upper bound for this WG

#if USE_TDM
  if (tid < 32) {
    tdm_load_2d_bf16(lds_off_of(&sK[0][0][0]), Kbase, HD_, S_, HD_, 32, HD_);
    tdm_load_2d_bf16(lds_off_of(&sV[0][0][0]), Vbase, S_, HD_, 32, HD_, S_);
  }
#endif

  for (int ch = 0; ch < nchunks; ++ch) {
    const int kv0 = ch * 32;
    const int buf = ch & 1;
#if USE_TDM
    if (tid < 32) {
      if (ch + 1 < nchunks) {
        const int kvn = kv0 + 32;
        tdm_load_2d_bf16(lds_off_of(&sK[buf ^ 1][0][0]), Kbase + (size_t)kvn * HD_,
                         HD_, S_, HD_, 32, HD_);
        tdm_load_2d_bf16(lds_off_of(&sV[buf ^ 1][0][0]), Vbase + kvn,
                         S_, HD_, 32, HD_, S_);
        __builtin_amdgcn_s_wait_tensorcnt(2);
      } else {
        __builtin_amdgcn_s_wait_tensorcnt(0);
      }
    }
#else
#pragma unroll
    for (int itc = 0; itc < 2; ++itc) {
      int c = tid + itc * 128;
      int row = c >> 3, pp = (c & 7) * 8;           // K: 32 rows x 64
      *(bf16x8*)&sK[buf][row][pp] =
          *(const bf16x8*)&Kbase[(size_t)(kv0 + row) * HD_ + pp];
      int vr = c >> 2, vp = (c & 3) * 8;            // V: 64 rows x 32
      *(bf16x8*)&sV[buf][vr][vp] =
          *(const bf16x8*)&Vbase[(size_t)vr * S_ + kv0 + vp];
    }
#endif
    __syncthreads();

    // scores: S(16x32) = Q(16x64) * K^T(64x32), two 16-col tiles, 2 WMMA each
    f32x8 sc[2];
#pragma unroll
    for (int nt = 0; nt < 2; ++nt) {
      f32x8 z = zero8();
      BFrag kb;
      int col = nt * 16 + r;
      kb.h[0] = *(const bf16x8*)&sK[buf][col][hi * 8];
      kb.h[1] = *(const bf16x8*)&sK[buf][col][16 + hi * 8];
      z = wmma_bf16(qa[0].v, kb.v, z);
      kb.h[0] = *(const bf16x8*)&sK[buf][col][32 + hi * 8];
      kb.h[1] = *(const bf16x8*)&sK[buf][col][48 + hi * 8];
      z = wmma_bf16(qa[1].v, kb.v, z);
      sc[nt] = z;
    }

    // online softmax over the 32 new kv columns
#pragma unroll
    for (int e = 0; e < 8; ++e) {
      int qrow = wq + e + 8 * hi;
      float a0 = sc[0][e] * SCALE_;
      float a1 = sc[1][e] * SCALE_;
      if (kv0 + r > qrow)      a0 = NEG_INF_;
      if (kv0 + 16 + r > qrow) a1 = NEG_INF_;
      float mx = fmaxf(a0, a1);
#pragma unroll
      for (int off = 1; off < 16; off <<= 1) mx = fmaxf(mx, __shfl_xor(mx, off));
      float mnew  = fmaxf(m_[e], mx);
      float alpha = __expf(m_[e] - mnew);
      float p0 = (a0 <= -1e29f) ? 0.f : __expf(a0 - mnew);
      float p1 = (a1 <= -1e29f) ? 0.f : __expf(a1 - mnew);
      float ps = p0 + p1;
#pragma unroll
      for (int off = 1; off < 16; off <<= 1) ps += __shfl_xor(ps, off);
      l_[e] = l_[e] * alpha + ps;
      m_[e] = mnew;
      alpha_[e] = alpha;
      sc[0][e] = p0;
      sc[1][e] = p1;
    }
#pragma unroll
    for (int ot = 0; ot < 4; ++ot)
#pragma unroll
      for (int e = 0; e < 8; ++e) o[ot][e] *= alpha_[e];

    // P (C-layout) -> per-wave LDS -> reload as A-fragment (16x32 bf16)
#pragma unroll
    for (int e = 0; e < 8; ++e) {
      int rr = e + 8 * hi;
      sP[w][rr][r]      = (__bf16)sc[0][e];
      sP[w][rr][16 + r] = (__bf16)sc[1][e];
    }
    __builtin_amdgcn_wave_barrier();   // keep ds_store before ds_load (in-order per wave)
    BFrag pa;
    pa.h[0] = *(const bf16x8*)&sP[w][r][hi * 8];
    pa.h[1] = *(const bf16x8*)&sP[w][r][16 + hi * 8];

    // O(16x64) += P(16x32) * V(32x64): four 16-col tiles
#pragma unroll
    for (int ot = 0; ot < 4; ++ot) {
      BFrag vb;
      int hd = ot * 16 + r;
      vb.h[0] = *(const bf16x8*)&sV[buf][hd][hi * 8];
      vb.h[1] = *(const bf16x8*)&sV[buf][hd][16 + hi * 8];
      o[ot] = wmma_bf16(pa.v, vb.v, o[ot]);
    }
    __syncthreads();
  }

  // epilogue: normalize and write (B,S,HID) as bf16 for the Wo GEMM
#pragma unroll
  for (int ot = 0; ot < 4; ++ot)
#pragma unroll
    for (int e = 0; e < 8; ++e) {
      int qrow = wq + e + 8 * hi;
      float val = o[ot][e] / fmaxf(l_[e], 1e-20f);
      O[((size_t)(b * S_ + qrow) * HID_) + hq * HD_ + ot * 16 + r] = (__bf16)val;
    }
}

// ---------- host launch ----------
extern "C" void kernel_launch(void* const* d_in, const int* in_sizes, int n_in,
                              void* d_out, int out_size, void* d_ws, size_t ws_size,
                              hipStream_t stream) {
  const float* hs   = (const float*)d_in[0];
  const float* cosT = (const float*)d_in[1];
  const float* sinT = (const float*)d_in[2];
  const float* Wq   = (const float*)d_in[3];
  const float* Wk   = (const float*)d_in[4];
  const float* Wv   = (const float*)d_in[5];
  const float* Wo   = (const float*)d_in[6];

  const int nHS = M_ * HID_;        // 4,194,304
  const int nWq = HID_ * HID_;      // 4,194,304
  const int nWk = HID_ * KVD_;      // 1,048,576
  const int nQ  = M_ * HQ_ * HD_;   // 4,194,304
  const int nK  = M_ * HKV_ * HD_;  // 1,048,576

  char* p = (char*)d_ws;
  __bf16* hsb  = (__bf16*)p;  p += (size_t)nHS * 2;
  __bf16* wqT  = (__bf16*)p;  p += (size_t)nWq * 2;   // transposed: N x K
  __bf16* wkT  = (__bf16*)p;  p += (size_t)nWk * 2;
  __bf16* wvT  = (__bf16*)p;  p += (size_t)nWk * 2;
  __bf16* woT  = (__bf16*)p;  p += (size_t)nWq * 2;
  float*  qraw = (float*)p;   p += (size_t)nQ * 4;
  float*  kraw = (float*)p;   p += (size_t)nK * 4;
  float*  vraw = (float*)p;   p += (size_t)nK * 4;
  __bf16* qb   = (__bf16*)p;  p += (size_t)nQ * 2;    // (B,HQ,S,HD)
  __bf16* kb   = (__bf16*)p;  p += (size_t)nK * 2;    // (B,HKV,S,HD)
  __bf16* vb   = (__bf16*)p;  p += (size_t)nK * 2;    // (B,HKV,HD,S)
  __bf16* attb = (__bf16*)p;  p += (size_t)nHS * 2;   // (B,S,HID)

  // 1) f32 -> bf16 conversions (weights transposed to N-major for TDM staging)
  cvt_f32_bf16<<<nHS / 256, 256, 0, stream>>>(hs, hsb, nHS);
  cvt_transpose_bf16<<<nWq / 256, 256, 0, stream>>>(Wq, wqT, HID_, HID_);
  cvt_transpose_bf16<<<nWk / 256, 256, 0, stream>>>(Wk, wkT, HID_, KVD_);
  cvt_transpose_bf16<<<nWk / 256, 256, 0, stream>>>(Wv, wvT, HID_, KVD_);
  cvt_transpose_bf16<<<nWq / 256, 256, 0, stream>>>(Wo, woT, HID_, HID_);

  // 2) QKV projections (TDM-staged WMMA GEMMs)
  gemm_bf16_f32<<<dim3(HID_ / BN, M_ / BM), 256, 0, stream>>>(hsb, wqT, qraw,
                                                              M_, HID_, HID_);
  gemm_bf16_f32<<<dim3(KVD_ / BN, M_ / BM), 256, 0, stream>>>(hsb, wkT, kraw,
                                                              M_, KVD_, HID_);
  gemm_bf16_f32<<<dim3(KVD_ / BN, M_ / BM), 256, 0, stream>>>(hsb, wvT, vraw,
                                                              M_, KVD_, HID_);

  // 3) RoPE (q,k) into head-major layouts; v cast+transpose
  rope_kernel<<<nQ / 256, 256, 0, stream>>>(qraw, qb, cosT, sinT, nQ, HQ_);
  rope_kernel<<<nK / 256, 256, 0, stream>>>(kraw, kb, cosT, sinT, nK, HKV_);
  cvt_v_transpose<<<nK / 256, 256, 0, stream>>>(vraw, vb, nK);

  // 4) causal flash attention (TDM-staged K/V, WMMA for QK^T and PV)
  attn_kernel<<<dim3(S_ / 64, HQ_, B_), 128, 0, stream>>>(qb, kb, vb, attb);

  // 5) output projection -> f32 d_out
  gemm_bf16_f32<<<dim3(HID_ / BN, M_ / BM), 256, 0, stream>>>(attb, woT,
                                                              (float*)d_out,
                                                              M_, HID_, HID_);
}
